// SG_7533372638046
// MI455X (gfx1250) — matrix operand order repaired and checked by reading, not compiled, not run on hardware
//
#include <hip/hip_runtime.h>
#include <hip/hip_bf16.h>

// ---------------------------------------------------------------------------
// PointNet++ set-abstraction for MI455X (gfx1250, wave32)
//   B=16, N=2048, Din=64, Dout=128, S=512, K=32
// GEMMs on v_wmma_f32_16x16x32_bf16 (f32 accum), fused recompute, LDS operands
// in fragment-native order (all WMMA inputs via ds_load_b128). Weights staged
// to LDS with the CDNA5 async Global->LDS copy path when available.
// ---------------------------------------------------------------------------

#define NPTS   2048
#define NBATCH 16
#define SCENT  512
#define KSAMP  32
#define DIN    64
#define DOUT   128
#define NGROUP (NBATCH * SCENT)          // 8192
#define NTOT_F 262144.0f                 // 8192 * 32 samples per channel
#define XYZ_OUT_ELEMS (NBATCH * SCENT * 3)
#define WBYTES (DOUT * 2 * DIN * 2)      // 32KB of bf16 per weight matrix

typedef __attribute__((ext_vector_type(16))) __bf16 v16bf;
typedef __attribute__((ext_vector_type(8)))  __bf16 v8bf;
typedef __attribute__((ext_vector_type(8)))  float  v8f;
typedef __attribute__((ext_vector_type(4)))  int    v4i;

#if defined(__has_builtin)
#  if __has_builtin(__builtin_amdgcn_global_load_async_to_lds_b128)
#    define SG_ASYNC_LDS 1
#  endif
#endif
#ifndef SG_ASYNC_LDS
#  define SG_ASYNC_LDS 0
#endif

#if SG_ASYNC_LDS
typedef __attribute__((address_space(1))) v4i as1_v4i;   // global
typedef __attribute__((address_space(3))) v4i as3_v4i;   // LDS
__device__ inline void async_copy_b128(const void* gsrc, void* ldst) {
  __builtin_amdgcn_global_load_async_to_lds_b128(
      (as1_v4i*)(const v4i*)gsrc, (as3_v4i*)(v4i*)ldst, 0, 0);
}
#endif

__device__ inline __bf16 f2bf(float f) {
  union { float f; unsigned u; } c; c.f = f;
  unsigned r = c.u + 0x7FFFu + ((c.u >> 16) & 1u);   // round-to-nearest-even
  unsigned short h = (unsigned short)(r >> 16);
  return __builtin_bit_cast(__bf16, h);
}

__device__ inline v16bf cat8(v8bf lo, v8bf hi) {
  return __builtin_shufflevector(lo, hi, 0, 1, 2, 3, 4, 5, 6, 7,
                                 8, 9, 10, 11, 12, 13, 14, 15);
}

// ------------------------- zero stats -------------------------------------
__global__ void zero_kernel(float* p, int n) {
  int i = blockIdx.x * blockDim.x + threadIdx.x;
  if (i < n) p[i] = 0.0f;
}

// ------------------------- weight f32 -> bf16 pre-convert ------------------
__global__ void conv_weights(const float* __restrict__ W1, const float* __restrict__ W2,
                             __bf16* __restrict__ w1bf, __bf16* __restrict__ w2bf) {
  int i = blockIdx.x * 256 + threadIdx.x;
  if (i < DOUT * 2 * DIN) { w1bf[i] = f2bf(W1[i]); w2bf[i] = f2bf(W2[i]); }
}

// ------------------------- farthest point sampling ------------------------
__global__ __launch_bounds__(256) void fps_kernel(const float* __restrict__ coords,
                                                  int* __restrict__ fidx,
                                                  float* __restrict__ out_xyz) {
  __shared__ float cx[NPTS], cy[NPTS], cz[NPTS], dist[NPTS];
  __shared__ float rv[256];
  __shared__ int   ri[256];
  int b = blockIdx.x, t = threadIdx.x;
  const float* cb = coords + (size_t)b * NPTS * 3;
  for (int i = t; i < NPTS; i += 256) {
    cx[i] = cb[i * 3 + 0]; cy[i] = cb[i * 3 + 1]; cz[i] = cb[i * 3 + 2];
    dist[i] = 1e10f;
  }
  __syncthreads();
  int far = 0;
  for (int s = 0; s < SCENT; ++s) {
    if (t == 0) {
      fidx[b * SCENT + s] = far;
      float* o = out_xyz + (size_t)(b * SCENT + s) * 3;
      o[0] = cx[far]; o[1] = cy[far]; o[2] = cz[far];
    }
    float fx = cx[far], fy = cy[far], fz = cz[far];
    float bv = -1.0f; int bi = 0;
    for (int i = t; i < NPTS; i += 256) {
      float dx = cx[i] - fx, dy = cy[i] - fy, dz = cz[i] - fz;
      float d  = dx * dx + dy * dy + dz * dz;
      float nd = fminf(dist[i], d); dist[i] = nd;
      if (nd > bv) { bv = nd; bi = i; }      // strict > keeps smallest index on tie
    }
    rv[t] = bv; ri[t] = bi;
    __syncthreads();
    for (int off = 128; off > 0; off >>= 1) {
      if (t < off) {
        float v2 = rv[t + off]; int i2 = ri[t + off];
        if (v2 > rv[t] || (v2 == rv[t] && i2 < ri[t])) { rv[t] = v2; ri[t] = i2; }
      }
      __syncthreads();
    }
    far = ri[0];
    __syncthreads();
  }
}

// ------------------------- gather centroid features -----------------------
__global__ void gather_newfeat(const float* __restrict__ x,
                               const int* __restrict__ fidx,
                               float* __restrict__ newfeat) {
  int i = blockIdx.x * 256 + threadIdx.x;
  if (i >= NGROUP * DIN) return;
  int g = i >> 6, c = i & 63;
  int b = g >> 9;
  int n = fidx[g];
  newfeat[i] = x[((size_t)b * DIN + c) * NPTS + n];
}

// ------------------------- ball query -------------------------------------
// One wave32 per centroid; each block's 8 centroids share a batch, so the
// batch's coords are staged once in LDS (24KB) and scanned from there.
__global__ __launch_bounds__(256) void qbp_kernel(const float* __restrict__ coords,
                                                  const float* __restrict__ new_xyz,
                                                  int* __restrict__ gidx) {
  __shared__ float cxl[NPTS], cyl[NPTS], czl[NPTS];
  const int t = threadIdx.x;
  const int b = (blockIdx.x * 8) >> 9;          // all 8 waves: same batch
  const float* cb = coords + (size_t)b * NPTS * 3;
  for (int i = t; i < NPTS; i += 256) {
    cxl[i] = cb[i * 3 + 0]; cyl[i] = cb[i * 3 + 1]; czl[i] = cb[i * 3 + 2];
  }
  __syncthreads();

  const int wave = blockIdx.x * 8 + (t >> 5);
  const int lane = t & 31;
  const float* q = new_xyz + (size_t)wave * 3;
  const float qx = q[0], qy = q[1], qz = q[2];
  int* gout = gidx + (size_t)wave * KSAMP;
  int count = 0, first = -1;
  for (int n0 = 0; n0 < NPTS && count < KSAMP; n0 += 32) {
    int n = n0 + lane;
    float dx = cxl[n] - qx, dy = cyl[n] - qy, dz = czl[n] - qz;
    bool pred = (dx * dx + dy * dy + dz * dz) <= 0.25f;   // RADIUS^2
    unsigned mask = (unsigned)__ballot(pred);
    if (first < 0 && mask) first = n0 + __ffs((int)mask) - 1;
    int pos = count + __popc(mask & ((1u << lane) - 1u));
    if (pred && pos < KSAMP) gout[pos] = n;
    count += __popc(mask);
  }
  if (count < KSAMP) {
    int fillIdx = (first >= 0) ? first : (NPTS - 1);
    int i = count + lane;
    if (i < KSAMP) gout[i] = fillIdx;
  }
}

// ------------------------- fused group MLP phases -------------------------
// PHASE 1: GEMM1, accumulate global sum/sumsq of h1        -> stats
// PHASE 2: GEMM1 -> BN1/ReLU -> GEMM2, accumulate h2 stats -> stats
// PHASE 3: full chain -> BN2/ReLU -> maxpool over K        -> d_out
template <int PHASE>
__global__ __launch_bounds__(128) void sg_phase_kernel(
    const float* __restrict__ x,
    const __bf16* __restrict__ w1bf, const __bf16* __restrict__ w2bf,
    const float* __restrict__ gamma1, const float* __restrict__ beta1,
    const float* __restrict__ gamma2, const float* __restrict__ beta2,
    const int*   __restrict__ gidx,
    const float* __restrict__ newfeat,
    float* __restrict__ stats,        // [256] sum | sumsq   (phase 1/2)
    const float* __restrict__ bn1,    // [256] mean | invstd (phase 2/3)
    const float* __restrict__ bn2,    // [256] mean | invstd (phase 3)
    float* __restrict__ out)          // d_out base          (phase 3)
{
  // Weights: row-major [out_ch][in_ch] (A-fragment order).
  __shared__ __align__(32) __bf16 w1l[DOUT * 2 * DIN];     // 32KB
  __shared__ __align__(32) __bf16 w2l[DOUT * DOUT];        // 32KB
  // Activations transposed: [col(k-sample)][in_ch] (B-fragment order).
  __shared__ __align__(32) __bf16 aggT[KSAMP * 2 * DIN];   // 8KB
  __shared__ __align__(32) __bf16 h1T[KSAMP * DOUT];       // 8KB
  __shared__ float  sumL[DOUT], sqL[DOUT];
  __shared__ float  sc1[DOUT], bi1[DOUT], sc2[DOUT], bi2[DOUT];
  __shared__ unsigned maxL[DOUT];
  __shared__ int gIdxL[KSAMP];

  const int tid  = threadIdx.x;
  const int wv   = tid >> 5;
  const int lane = tid & 31;
  const int half = lane >> 4;      // selects contiguous K-subrange k = 16*half + s
  const int lr   = lane & 15;

  // ---- stage bf16 weights into LDS (async Global->LDS path if available) --
#if SG_ASYNC_LDS
  {
    const char* s1 = (const char*)w1bf;
    const char* s2 = (const char*)w2bf;
    char* d1 = (char*)&w1l[0];
    char* d2 = (char*)&w2l[0];
    for (int off = tid * 16; off < WBYTES; off += 128 * 16) {
      async_copy_b128(s1 + off, d1 + off);
      if (PHASE >= 2) async_copy_b128(s2 + off, d2 + off);
    }
#  if __has_builtin(__builtin_amdgcn_s_wait_asynccnt)
    __builtin_amdgcn_s_wait_asynccnt(0);
#  else
    asm volatile("s_wait_asynccnt 0" ::: "memory");
#  endif
  }
#else
  for (int i = tid; i < DOUT * 2 * DIN; i += 128) {
    w1l[i] = w1bf[i];
    if (PHASE >= 2) w2l[i] = w2bf[i];
  }
#endif

  if (tid < DOUT) {
    if (PHASE != 3) { sumL[tid] = 0.0f; sqL[tid] = 0.0f; }
    if (PHASE >= 2) {
      float s = bn1[DOUT + tid] * gamma1[tid];
      sc1[tid] = s; bi1[tid] = beta1[tid] - bn1[tid] * s;
    }
    if (PHASE == 3) {
      float s = bn2[DOUT + tid] * gamma2[tid];
      sc2[tid] = s; bi2[tid] = beta2[tid] - bn2[tid] * s;
    }
  }
  __syncthreads();

  // Per-lane register accumulators for BN stats: channel set is invariant
  // across n-tiles and groups -> flush to LDS once at kernel end.
  float rs[16] = {0.f, 0.f, 0.f, 0.f, 0.f, 0.f, 0.f, 0.f,
                  0.f, 0.f, 0.f, 0.f, 0.f, 0.f, 0.f, 0.f};
  float rq[16] = {0.f, 0.f, 0.f, 0.f, 0.f, 0.f, 0.f, 0.f,
                  0.f, 0.f, 0.f, 0.f, 0.f, 0.f, 0.f, 0.f};

  for (int g = 0; g < 8; ++g) {
    const int group = blockIdx.x * 8 + g;
    const int b = group >> 9, s = group & 511;
    if (tid < KSAMP) gIdxL[tid] = gidx[(size_t)group * KSAMP + tid];
    if (PHASE == 3 && tid < DOUT) maxL[tid] = 0u;   // ReLU output >= 0
    __syncthreads();

    // Build aggT[k][c]: c<64 -> grouped-center, c>=64 -> center broadcast.
    const float* nf = newfeat + (size_t)group * DIN;
    for (int e = tid; e < KSAMP * 2 * DIN; e += 128) {
      int k = e >> 7, c = e & 127;
      float v;
      if (c < DIN) {
        int n = gIdxL[k]; n = min(max(n, 0), NPTS - 1);
        v = x[((size_t)b * DIN + c) * NPTS + n] - nf[c];
      } else {
        v = nf[c - DIN];
      }
      aggT[e] = f2bf(v);
    }
    __syncthreads();

    // ---------------- GEMM1: h1 = W1 (128x128) @ agg (128x32) -------------
#pragma unroll
    for (int mi = 0; mi < 2; ++mi) {
      const int mtile = wv * 2 + mi;
      v16bf a[4];
#pragma unroll
      for (int kk = 0; kk < 4; ++kk) {     // A 16x32 bf16 fragment layout
        const __bf16* wp = &w1l[(mtile * 16 + lr) * 128 + kk * 32 + half * 16];
        a[kk] = *(const v16bf*)wp;
      }
#pragma unroll
      for (int ntile = 0; ntile < 2; ++ntile) {
        v8f acc = {};
#pragma unroll
        for (int kk = 0; kk < 4; ++kk) {   // B 32x16 fragment: contiguous 32B
          v16bf bb = *(const v16bf*)&aggT[(ntile * 16 + lr) * 128 +
                                          kk * 32 + half * 16];
          acc = __builtin_amdgcn_wmma_f32_16x16x32_bf16(false, a[kk], false, bb,
                                                        (short)0, acc, false, false);
        }
        const int ch0 = mtile * 16 + half * 8;        // C/D fragment layout
        const int col = ntile * 16 + lr;
        if (PHASE == 1) {
#pragma unroll
          for (int j = 0; j < 8; ++j) {
            float v = acc[j];
            rs[mi * 8 + j] += v;
            rq[mi * 8 + j] += v * v;
          }
        } else {
          v8bf pk;
#pragma unroll
          for (int j = 0; j < 8; ++j) {
            float hv = fmaxf(acc[j] * sc1[ch0 + j] + bi1[ch0 + j], 0.0f);
            pk[j] = f2bf(hv);
          }
          *(v8bf*)&h1T[col * 128 + ch0] = pk;         // one ds_store_b128
        }
      }
    }
    __syncthreads();

    // ---------------- GEMM2: h2 = W2 (128x128) @ h1r (128x32) -------------
    if (PHASE >= 2) {
#pragma unroll
      for (int mi = 0; mi < 2; ++mi) {
        const int mtile = wv * 2 + mi;
        v16bf a[4];
#pragma unroll
        for (int kk = 0; kk < 4; ++kk) {
          const __bf16* wp = &w2l[(mtile * 16 + lr) * 128 + kk * 32 + half * 16];
          a[kk] = *(const v16bf*)wp;
        }
#pragma unroll
        for (int ntile = 0; ntile < 2; ++ntile) {
          v8f acc = {};
#pragma unroll
          for (int kk = 0; kk < 4; ++kk) {
            v16bf bb = *(const v16bf*)&h1T[(ntile * 16 + lr) * 128 +
                                           kk * 32 + half * 16];
            acc = __builtin_amdgcn_wmma_f32_16x16x32_bf16(false, a[kk], false, bb,
                                                          (short)0, acc, false, false);
          }
          const int ch0 = mtile * 16 + half * 8;
#pragma unroll
          for (int j = 0; j < 8; ++j) {
            float v = acc[j];
            if (PHASE == 2) {
              rs[mi * 8 + j] += v;
              rq[mi * 8 + j] += v * v;
            } else {
              float hv = fmaxf(v * sc2[ch0 + j] + bi2[ch0 + j], 0.0f);
              atomicMax(&maxL[ch0 + j], __float_as_uint(hv));  // valid: hv >= 0
            }
          }
        }
      }
      __syncthreads();
      if (PHASE == 3 && tid < DOUT) {
        out[XYZ_OUT_ELEMS + ((size_t)b * DOUT + tid) * SCENT + s] =
            __uint_as_float(maxL[tid]);
      }
      __syncthreads();
    }
  }

  // ---- flush register stat accumulators: regs -> LDS -> global ------------
  if (PHASE != 3) {
#pragma unroll
    for (int mi = 0; mi < 2; ++mi)
#pragma unroll
      for (int j = 0; j < 8; ++j) {
        const int ch = (wv * 2 + mi) * 16 + half * 8 + j;
        atomicAdd(&sumL[ch], rs[mi * 8 + j]);
        atomicAdd(&sqL[ch], rq[mi * 8 + j]);
      }
    __syncthreads();
    if (tid < DOUT) {
      atomicAdd(&stats[tid],        sumL[tid]);
      atomicAdd(&stats[DOUT + tid], sqL[tid]);
    }
  }
}

// ------------------------- BN stat finalize -------------------------------
__global__ void bn_finalize(const float* __restrict__ stats, float* __restrict__ bn) {
  int c = threadIdx.x;
  if (c < DOUT) {
    const float inv = 1.0f / NTOT_F;
    float m = stats[c] * inv;
    float v = stats[DOUT + c] * inv - m * m;
    bn[c]        = m;
    bn[DOUT + c] = rsqrtf(v + 1e-5f);
  }
}

// ---------------------------------------------------------------------------
extern "C" void kernel_launch(void* const* d_in, const int* in_sizes, int n_in,
                              void* d_out, int out_size, void* d_ws, size_t ws_size,
                              hipStream_t stream) {
  const float* x      = (const float*)d_in[0];
  const float* coords = (const float*)d_in[1];
  const float* W1     = (const float*)d_in[2];
  const float* W2     = (const float*)d_in[3];
  const float* gamma1 = (const float*)d_in[4];
  const float* beta1  = (const float*)d_in[5];
  const float* gamma2 = (const float*)d_in[6];
  const float* beta2  = (const float*)d_in[7];
  float* out = (float*)d_out;

  // workspace layout (~3.3 MB)
  int*    fidx    = (int*)d_ws;                       // 8192
  int*    gidx    = fidx + NGROUP;                    // 8192*32
  float*  newfeat = (float*)(gidx + NGROUP * KSAMP);  // 8192*64
  float*  stats1  = newfeat + (size_t)NGROUP * DIN;   // 256
  float*  stats2  = stats1 + 2 * DOUT;                // 256
  float*  bn1     = stats2 + 2 * DOUT;                // 256
  float*  bn2     = bn1 + 2 * DOUT;                   // 256
  __bf16* w1bf    = (__bf16*)(bn2 + 2 * DOUT);        // 16384 bf16 (32KB)
  __bf16* w2bf    = w1bf + DOUT * 2 * DIN;            // 16384 bf16 (32KB)

  zero_kernel<<<2, 256, 0, stream>>>(stats1, 4 * DOUT);
  conv_weights<<<(DOUT * 2 * DIN + 255) / 256, 256, 0, stream>>>(W1, W2, w1bf, w2bf);
  fps_kernel<<<NBATCH, 256, 0, stream>>>(coords, fidx, out);
  gather_newfeat<<<(NGROUP * DIN + 255) / 256, 256, 0, stream>>>(x, fidx, newfeat);
  qbp_kernel<<<NGROUP / 8, 256, 0, stream>>>(coords, out, gidx);

  sg_phase_kernel<1><<<NGROUP / 8, 128, 0, stream>>>(
      x, w1bf, w2bf, gamma1, beta1, gamma2, beta2, gidx, newfeat, stats1, bn1, bn2, out);
  bn_finalize<<<1, 128, 0, stream>>>(stats1, bn1);
  sg_phase_kernel<2><<<NGROUP / 8, 128, 0, stream>>>(
      x, w1bf, w2bf, gamma1, beta1, gamma2, beta2, gidx, newfeat, stats2, bn1, bn2, out);
  bn_finalize<<<1, 128, 0, stream>>>(stats2, bn2);
  sg_phase_kernel<3><<<NGROUP / 8, 128, 0, stream>>>(
      x, w1bf, w2bf, gamma1, beta1, gamma2, beta2, gidx, newfeat, stats2, bn1, bn2, out);
}
